// StructureModule_9354438771229
// MI455X (gfx1250) — compile-verified, matrix operand order checked by reading, not run on hardware
//
#include <hip/hip_runtime.h>
#include <hip/hip_bf16.h>
#include <math.h>

// ---------------------------------------------------------------------------
// IPA / StructureModule for MI455X (gfx1250), wave32 + WMMA f16.
// All matmuls go through v_wmma_f32_16x16x32_f16 (fp32 accum).
// Memory roofline: 2 passes over inputs_2d (604 MB) dominate -> staging is
// float4/b128 + ds_store_b64, double-buffered, load/store phases split so
// multiple global loads are in flight before the first s_wait_loadcnt.
// ---------------------------------------------------------------------------

#define NRES 768
#define NH   12
#define SQK  16
#define SV   16
#define PQK  4
#define PV   8
#define C1   384
#define C2   128
#define OUTD 384
#define FDIM 2112          // H*(C2+SV+4*PV)
#define SW_  0.14433756729740643f   // sqrt(1/(3*16))
#define PW_  0.13608276348795434f   // sqrt(1/(3*4*4.5))
#define AW_  0.5773502691896258f    // sqrt(1/3)
#define EPSD 1e-8f

typedef __attribute__((ext_vector_type(16))) _Float16 v16h;
typedef __attribute__((ext_vector_type(4)))  _Float16 v4h;
typedef __attribute__((ext_vector_type(8)))  float    v8f;

__device__ __forceinline__ v8f wmma_f16(v16h a, v16h b, v8f c) {
  return __builtin_amdgcn_wmma_f32_16x16x32_f16(
      /*neg_a=*/false, a, /*neg_b=*/false, b,
      /*c_mod=*/(short)0, c, /*reuse_a=*/false, /*reuse_b=*/false);
}

__device__ __forceinline__ v4h cvt4(float4 v) {
  v4h h = {(_Float16)v.x, (_Float16)v.y, (_Float16)v.z, (_Float16)v.w};
  return h;
}

// Gather a 16x32 (MxK) f16 operand tile from LDS (row-major, stride ld halves)
// into the CDNA5 WMMA A/B VGPR striping:
//   lanes 0-15 : row m=lane,    K in {0..7, 16..23}
//   lanes 16-31: row m=lane-16, K in {8..15, 24..31}
// Contiguous 8-half groups -> compiler emits 2x ds_load_b128 per operand.
__device__ __forceinline__ v16h tile_lds(const _Float16* src, int ld) {
  const int lane = threadIdx.x & 31;
  const int m  = lane & 15;
  const int kb = (lane >> 4) << 3;       // 0 or 8
  const _Float16* row = src + m * ld;
  v16h r;
#pragma unroll
  for (int j = 0; j < 16; ++j) {
    int k = kb + (j < 8 ? j : j + 8);
    r[j] = row[k];
  }
  return r;
}

// Same gather straight from global fp32 via 4x float4 (b128) loads.
__device__ __forceinline__ v16h tile_f32g(const float* src, int ld) {
  const int lane = threadIdx.x & 31;
  const int m  = lane & 15;
  const int kb = (lane >> 4) << 3;
  const float* row = src + (long long)m * ld + kb;
  float4 p0 = *(const float4*)(row + 0);
  float4 p1 = *(const float4*)(row + 4);
  float4 p2 = *(const float4*)(row + 16);
  float4 p3 = *(const float4*)(row + 20);
  v16h r;
  r[0]  = (_Float16)p0.x; r[1]  = (_Float16)p0.y; r[2]  = (_Float16)p0.z; r[3]  = (_Float16)p0.w;
  r[4]  = (_Float16)p1.x; r[5]  = (_Float16)p1.y; r[6]  = (_Float16)p1.z; r[7]  = (_Float16)p1.w;
  r[8]  = (_Float16)p2.x; r[9]  = (_Float16)p2.y; r[10] = (_Float16)p2.z; r[11] = (_Float16)p2.w;
  r[12] = (_Float16)p3.x; r[13] = (_Float16)p3.y; r[14] = (_Float16)p3.z; r[15] = (_Float16)p3.w;
  return r;
}

// ---------------------------------------------------------------------------
// Generic batched GEMM:  C[z] = A[z] (MxK, fp32) * B[z] (KxN, fp32) + bias
// f16 WMMA compute, fp32 accumulate.  Block = 128 thr (4 waves) = 64x16 tile.
// REQUIREMENTS (all call sites satisfy): K % 32 == 0, M % 64 == 0,
// A rows/base 16B-aligned.  Only N may be ragged (attn_2d: N=12).
// Double-buffered LDS: one barrier per K-step; staging issues all 8 global
// loads before any LDS store so they overlap in the memory system.
// ---------------------------------------------------------------------------
__global__ void gemm_wmma(const float* __restrict__ A, int lda, long long strideA,
                          const float* __restrict__ B, int ldb, long long strideB,
                          const float* __restrict__ bias,
                          float* __restrict__ C, int ldc, long long strideC,
                          int M, int Nc, int K) {
  __shared__ _Float16 sA[2][64 * 32];   // 64x32 A tiles
  __shared__ _Float16 sBt[2][16 * 32];  // B tiles, transposed to NxK

  const int z = blockIdx.z;
  A += (long long)z * strideA;
  B += (long long)z * strideB;
  C += (long long)z * strideC;

  const int m0   = blockIdx.x * 64;
  const int n0   = blockIdx.y * 16;
  const int tid  = threadIdx.x;       // 0..127
  const int wave = tid >> 5;

  auto stage = [&](int buf, int k0) {
    // ---- load phase: 4x b128 (A) + 4x b32 (B) all in flight ----
    float4 va[4];
#pragma unroll
    for (int i = 0; i < 4; ++i) {
      int idx4 = tid + i * 128;          // 0..511
      int r = idx4 >> 3, c4 = idx4 & 7;  // row, float4-within-row
      va[i] = *(const float4*)(A + (long long)(m0 + r) * lda + k0 + c4 * 4);
    }
    float vb[4];
#pragma unroll
    for (int i = 0; i < 4; ++i) {
      int idx = tid + i * 128;           // 0..511
      int kk = idx >> 4, nn = idx & 15;
      int gn = n0 + nn;
      int gns = (gn < Nc) ? gn : (Nc - 1);
      vb[i] = B[(long long)(k0 + kk) * ldb + gns];
    }
    // ---- store phase: convert + ds_store_b64 / b16 ----
#pragma unroll
    for (int i = 0; i < 4; ++i) {
      int idx4 = tid + i * 128;
      int r = idx4 >> 3, c4 = idx4 & 7;
      *(v4h*)(&sA[buf][r * 32 + c4 * 4]) = cvt4(va[i]);
    }
#pragma unroll
    for (int i = 0; i < 4; ++i) {
      int idx = tid + i * 128;
      int kk = idx >> 4, nn = idx & 15;
      int gn = n0 + nn;
      float v = (gn < Nc) ? vb[i] : 0.f;
      sBt[buf][nn * 32 + kk] = (_Float16)v;
    }
  };

  v8f acc = {0.f, 0.f, 0.f, 0.f, 0.f, 0.f, 0.f, 0.f};
  stage(0, 0);
  int buf = 0;
  for (int k0 = 0; k0 < K; k0 += 32) {
    __syncthreads();
    if (k0 + 32 < K) stage(buf ^ 1, k0 + 32);
    v16h a = tile_lds(sA[buf] + wave * 16 * 32, 32);
    v16h b = tile_lds(sBt[buf], 32);
    acc = wmma_f16(a, b, acc);
    buf ^= 1;
  }

  // store 16x16 (C/D layout: lane n = col, VGPR r -> row mb+r); M % 64 == 0
  const int lane = tid & 31;
  const int n  = lane & 15;
  const int mb = (lane >> 4) << 3;
  const int gn = n0 + n;
  if (gn < Nc) {
    float bv = bias ? bias[gn] : 0.f;
    const long long base = (long long)(m0 + wave * 16 + mb) * ldc + gn;
#pragma unroll
    for (int r = 0; r < 8; ++r)
      C[base + (long long)r * ldc] = acc[r] + bv;
  }
}

// ---------------------------------------------------------------------------
// Prep: apply frames, build extended Q/K (28-dim: 16 scalar + 12 point, point
// weight folded into Q), point-norm rank-1 terms, and combined V (16+24=40).
// One thread per (h, n).
// ---------------------------------------------------------------------------
__global__ void prep_kernel(const float* __restrict__ q_scalar,   // 768x192
                            const float* __restrict__ kv_scalar,  // 768x384
                            const float* __restrict__ qp_l,       // 768x144
                            const float* __restrict__ kvp_l,      // 768x432
                            const float* __restrict__ rot,        // 768x9
                            const float* __restrict__ trans,      // 768x3
                            const float* __restrict__ tpw,        // 12
                            float* __restrict__ Qe,               // [h][n][32]
                            float* __restrict__ Ke,               // [h][n][32]
                            float* __restrict__ Qn2,              // [h][n]
                            float* __restrict__ Kn2,              // [h][n]
                            float* __restrict__ Vc,               // [h][n][48]
                            float* __restrict__ pwbuf)            // 12
{
  int t = blockIdx.x * blockDim.x + threadIdx.x;
  if (t >= NRES * NH) return;
  int h = t / NRES, n = t % NRES;

  float x = tpw[h];
  float pw = ((x > 20.f) ? x : log1pf(__expf(x))) * PW_;   // softplus * PW
  if (n == 0) pwbuf[h] = pw;

  float R[9], T[3];
#pragma unroll
  for (int i = 0; i < 9; ++i) R[i] = rot[n * 9 + i];
#pragma unroll
  for (int i = 0; i < 3; ++i) T[i] = trans[n * 3 + i];

  long long qi = ((long long)h * NRES + n) * 32;
  long long vi = ((long long)h * NRES + n) * 48;

  // scalar parts
#pragma unroll
  for (int d = 0; d < 16; ++d) {
    Qe[qi + d] = SW_ * q_scalar[(long long)n * 192 + h * 16 + d];
    Ke[qi + d] = kv_scalar[(long long)n * 384 + h * 32 + d];
    Vc[vi + d] = kv_scalar[(long long)n * 384 + h * 32 + 16 + d];
  }

  // q points: global frame, fold pw into Q side
  float qn2 = 0.f;
#pragma unroll
  for (int c = 0; c < 3; ++c) {
#pragma unroll
    for (int p = 0; p < PQK; ++p) {
      float g = T[c];
#pragma unroll
      for (int j = 0; j < 3; ++j)
        g += R[c * 3 + j] * qp_l[(long long)n * 144 + j * 48 + h * PQK + p];
      Qe[qi + 16 + c * PQK + p] = pw * g;
      qn2 += g * g;
    }
  }
  Qe[qi + 28] = 0.f; Qe[qi + 29] = 0.f; Qe[qi + 30] = 0.f; Qe[qi + 31] = 0.f;
  Qn2[(long long)h * NRES + n] = qn2;

  // k/v points: global frame
  float kn2 = 0.f;
#pragma unroll
  for (int c = 0; c < 3; ++c) {
#pragma unroll
    for (int p = 0; p < (PQK + PV); ++p) {
      float g = T[c];
#pragma unroll
      for (int j = 0; j < 3; ++j)
        g += R[c * 3 + j] * kvp_l[(long long)n * 432 + j * 144 + h * 12 + p];
      if (p < PQK) { Ke[qi + 16 + c * PQK + p] = g; kn2 += g * g; }
      else           Vc[vi + 16 + c * PV + (p - PQK)] = g;
    }
  }
  Ke[qi + 28] = 0.f; Ke[qi + 29] = 0.f; Ke[qi + 30] = 0.f; Ke[qi + 31] = 0.f;
#pragma unroll
  for (int d = 40; d < 48; ++d) Vc[vi + d] = 0.f;
  Kn2[(long long)h * NRES + n] = kn2;
}

// ---------------------------------------------------------------------------
// Fused attention logits + softmax.
// grid = (48 i-tiles, 12 heads), block = 256 (8 waves).
// Each wave does 6 j-tiles of 16 via one WMMA (K=32 covers the 28 live dims),
// adds the rank-1 point-norm terms + attn_2d*AW + mask, then the block does a
// row softmax over the 16x768 logits held in LDS and writes attn (fp32).
// ---------------------------------------------------------------------------
__global__ void logits_softmax_kernel(const float* __restrict__ Qe,
                                      const float* __restrict__ Ke,
                                      const float* __restrict__ Qn2,
                                      const float* __restrict__ Kn2,
                                      const float* __restrict__ attn2d,  // [i*N+j][12]
                                      const float* __restrict__ pwbuf,
                                      const float* __restrict__ mask,    // [N]
                                      float* __restrict__ attn)          // [h][i][j]
{
  __shared__ float sL[16 * NRES];   // 48 KB

  const int h    = blockIdx.y;
  const int i0   = blockIdx.x * 16;
  const int tid  = threadIdx.x;     // 0..255
  const int wave = tid >> 5;
  const int lane = tid & 31;
  const float pw = pwbuf[h];

  const float* Qh = Qe + (long long)h * NRES * 32;
  const float* Kh = Ke + (long long)h * NRES * 32;

  // A operand: 16 query rows, 32-dim extended channel (single K step)
  v16h a = tile_f32g(Qh + (long long)i0 * 32, 32);

  const int nn = lane & 15;
  const int mb = (lane >> 4) << 3;

  // hoist per-output-row constants (same for all 6 j-tiles)
  float qrow[8], mrow[8];
#pragma unroll
  for (int r = 0; r < 8; ++r) {
    int ii = i0 + mb + r;
    qrow[r] = Qn2[(long long)h * NRES + ii];
    mrow[r] = mask[ii];
  }

  for (int t = 0; t < 6; ++t) {
    int j0 = (wave * 6 + t) * 16;
    v16h b = tile_f32g(Kh + (long long)j0 * 32, 32);
    v8f d = {0.f, 0.f, 0.f, 0.f, 0.f, 0.f, 0.f, 0.f};
    d = wmma_f16(a, b, d);

    int jj = j0 + nn;
    float kn = Kn2[(long long)h * NRES + jj];
    float mj = mask[jj];
#pragma unroll
    for (int r = 0; r < 8; ++r) {
      int ii = i0 + mb + r;
      float v = d[r]
              - 0.5f * pw * (qrow[r] + kn)
              + AW_ * attn2d[((long long)ii * NRES + jj) * NH + h]
              - 100000.f * (1.f - mrow[r] * mj);
      sL[(mb + r) * NRES + jj] = v;
    }
  }
  __syncthreads();

  // softmax: wave handles rows 2*wave, 2*wave+1
  for (int rr = 0; rr < 2; ++rr) {
    int row = wave * 2 + rr;
    float* L = sL + row * NRES;
    float mx = -1e30f;
    for (int j = lane; j < NRES; j += 32) mx = fmaxf(mx, L[j]);
#pragma unroll
    for (int o = 16; o; o >>= 1) mx = fmaxf(mx, __shfl_xor(mx, o, 32));
    float s = 0.f;
    for (int j = lane; j < NRES; j += 32) { float e = __expf(L[j] - mx); L[j] = e; s += e; }
#pragma unroll
    for (int o = 16; o; o >>= 1) s += __shfl_xor(s, o, 32);
    float inv = 1.f / s;
    float* dst = attn + ((long long)h * NRES + (i0 + row)) * NRES;
    for (int j = lane; j < NRES; j += 32) dst[j] = L[j] * inv;
  }
}

// ---------------------------------------------------------------------------
// attn_over_2d: for each residue i, (12x768 attn rows) @ (768x128 pair slab),
// written straight into final_act columns [576, 2112).
// grid = 768, block = 256 (8 waves, one 16-col slice each).
// Second (and last) full read of inputs_2d: float4-coalesced, double-buffered,
// loads issued before stores for memory-level parallelism.
// ---------------------------------------------------------------------------
__global__ void attn_over_2d_kernel(const float* __restrict__ attn,   // [h][i][j]
                                    const float* __restrict__ in2d,   // [i][j][128]
                                    float* __restrict__ final_act)    // [i][2112]
{
  __shared__ _Float16 sA[2][16 * 32];    // heads x k
  __shared__ _Float16 sB[2][32 * C2];    // k x channels

  const int i    = blockIdx.x;
  const int tid  = threadIdx.x;       // 0..255
  const int wave = tid >> 5;
  const int lane = tid & 31;
  const int c0   = wave * 16;

  auto stage = [&](int buf, int k0) {
    // ---- load phase ----
    const float4* src = (const float4*)(in2d + ((long long)i * NRES + k0) * C2);
    float4 vb[4];
#pragma unroll
    for (int q = 0; q < 4; ++q) vb[q] = src[tid + q * 256];

    const int r = tid >> 3, c4 = tid & 7;
    const bool hasA = (tid < 128);      // per-wave uniform (waves 0-3)
    float4 va = {0.f, 0.f, 0.f, 0.f};
    if (hasA) {
      int rs = (r < NH) ? r : (NH - 1);
      va = *(const float4*)(attn + ((long long)rs * NRES + i) * NRES + k0 + c4 * 4);
    }
    // ---- store phase ----
#pragma unroll
    for (int q = 0; q < 4; ++q)
      *(v4h*)(&sB[buf][(tid + q * 256) * 4]) = cvt4(vb[q]);
    if (hasA) {
      if (r >= NH) { va.x = 0.f; va.y = 0.f; va.z = 0.f; va.w = 0.f; }
      *(v4h*)(&sA[buf][r * 32 + c4 * 4]) = cvt4(va);
    }
  };

  v8f acc = {0.f, 0.f, 0.f, 0.f, 0.f, 0.f, 0.f, 0.f};
  stage(0, 0);
  int buf = 0;
  for (int k0 = 0; k0 < NRES; k0 += 32) {
    __syncthreads();
    if (k0 + 32 < NRES) stage(buf ^ 1, k0 + 32);

    v16h a = tile_lds(sA[buf], 32);
    // B operand gather: Bt[n][k] = sB[k][c0+n]
    v16h b;
    {
      const int n2 = lane & 15;
      const int kb = (lane >> 4) << 3;
#pragma unroll
      for (int j = 0; j < 16; ++j) {
        int k = kb + (j < 8 ? j : j + 8);
        b[j] = sB[buf][k * C2 + c0 + n2];
      }
    }
    acc = wmma_f16(a, b, acc);
    buf ^= 1;
  }

  const int n  = lane & 15;
  const int mb = (lane >> 4) << 3;
#pragma unroll
  for (int r = 0; r < 8; ++r) {
    int hh = mb + r;
    if (hh < NH)
      final_act[(long long)i * FDIM + 576 + hh * C2 + c0 + n] = acc[r];
  }
}

// ---------------------------------------------------------------------------
// Assemble final_act cols [0,576): result_scalar, rpl (inverse frame), dist.
// One thread per (h, n).
// ---------------------------------------------------------------------------
__global__ void assemble_kernel(const float* __restrict__ resE,   // [h][n][48]
                                const float* __restrict__ rot,    // [n][9]
                                const float* __restrict__ trans,  // [n][3]
                                float* __restrict__ final_act)    // [n][2112]
{
  int t = blockIdx.x * blockDim.x + threadIdx.x;
  if (t >= NRES * NH) return;
  int h = t / NRES, n = t % NRES;

  const float* R  = rot + n * 9;
  const float* T  = trans + n * 3;
  const float* re = resE + ((long long)h * NRES + n) * 48;
  float* fa = final_act + (long long)n * FDIM;

#pragma unroll
  for (int d = 0; d < SV; ++d) fa[h * SV + d] = re[d];

#pragma unroll
  for (int p = 0; p < PV; ++p) {
    float g0 = re[16 + 0 * PV + p] - T[0];
    float g1 = re[16 + 1 * PV + p] - T[1];
    float g2 = re[16 + 2 * PV + p] - T[2];
    // rpl[i] = sum_j R[j][i] * g[j]   (inverse rotation)
    float l0 = R[0] * g0 + R[3] * g1 + R[6] * g2;
    float l1 = R[1] * g0 + R[4] * g1 + R[7] * g2;
    float l2 = R[2] * g0 + R[5] * g1 + R[8] * g2;
    fa[192 + 0 * 96 + h * PV + p] = l0;
    fa[192 + 1 * 96 + h * PV + p] = l1;
    fa[192 + 2 * 96 + h * PV + p] = l2;
    fa[480 + h * PV + p] = sqrtf(EPSD + l0 * l0 + l1 * l1 + l2 * l2);
  }
}

// ---------------------------------------------------------------------------
// Host launcher
// ---------------------------------------------------------------------------
extern "C" void kernel_launch(void* const* d_in, const int* in_sizes, int n_in,
                              void* d_out, int out_size, void* d_ws, size_t ws_size,
                              hipStream_t stream) {
  (void)in_sizes; (void)n_in; (void)out_size; (void)ws_size;

  const float* in1d  = (const float*)d_in[0];
  const float* in2d  = (const float*)d_in[1];
  const float* mask  = (const float*)d_in[2];
  const float* rot   = (const float*)d_in[3];
  const float* trans = (const float*)d_in[4];
  const float* wq    = (const float*)d_in[5];
  const float* bq    = (const float*)d_in[6];
  const float* wkv   = (const float*)d_in[7];
  const float* bkv   = (const float*)d_in[8];
  const float* wqp   = (const float*)d_in[9];
  const float* bqp   = (const float*)d_in[10];
  const float* wkvp  = (const float*)d_in[11];
  const float* bkvp  = (const float*)d_in[12];
  const float* tpw   = (const float*)d_in[13];
  const float* w2d   = (const float*)d_in[14];
  const float* b2d   = (const float*)d_in[15];
  const float* wout  = (const float*)d_in[16];
  const float* bout  = (const float*)d_in[17];
  float* out = (float*)d_out;

  // workspace layout (floats)
  float* p = (float*)d_ws;
  float* q_scalar  = p; p += (long long)NRES * 192;
  float* kv_scalar = p; p += (long long)NRES * 384;
  float* qp_l      = p; p += (long long)NRES * 144;
  float* kvp_l     = p; p += (long long)NRES * 432;
  float* Qe        = p; p += (long long)NH * NRES * 32;
  float* Ke        = p; p += (long long)NH * NRES * 32;
  float* Qn2       = p; p += (long long)NH * NRES;
  float* Kn2       = p; p += (long long)NH * NRES;
  float* pwbuf     = p; p += 16;
  float* Vc        = p; p += (long long)NH * NRES * 48;
  float* attn2d    = p; p += (long long)NRES * NRES * NH;
  float* attn      = p; p += (long long)NH * NRES * NRES;
  float* resE      = p; p += (long long)NH * NRES * 48;
  float* final_act = p; p += (long long)NRES * FDIM;

  // 1) input projections (K=384)
  gemm_wmma<<<dim3(12, 12, 1), 128, 0, stream>>>(
      in1d, C1, 0, wq, 192, 0, bq, q_scalar, 192, 0, NRES, 192, C1);
  gemm_wmma<<<dim3(12, 24, 1), 128, 0, stream>>>(
      in1d, C1, 0, wkv, 384, 0, bkv, kv_scalar, 384, 0, NRES, 384, C1);
  gemm_wmma<<<dim3(12, 9, 1), 128, 0, stream>>>(
      in1d, C1, 0, wqp, 144, 0, bqp, qp_l, 144, 0, NRES, 144, C1);
  gemm_wmma<<<dim3(12, 27, 1), 128, 0, stream>>>(
      in1d, C1, 0, wkvp, 432, 0, bkvp, kvp_l, 432, 0, NRES, 432, C1);

  // 2) frames + extended Q/K/V construction
  prep_kernel<<<(NRES * NH + 255) / 256, 256, 0, stream>>>(
      q_scalar, kv_scalar, qp_l, kvp_l, rot, trans, tpw,
      Qe, Ke, Qn2, Kn2, Vc, pwbuf);

  // 3) attn_2d = inputs_2d @ w_attn2d  (first 302 MB pass over inputs_2d)
  gemm_wmma<<<dim3((NRES * NRES) / 64, 1, 1), 128, 0, stream>>>(
      in2d, C2, 0, w2d, NH, 0, b2d, attn2d, NH, 0, NRES * NRES, NH, C2);

  // 4) fused logits + softmax -> attn
  logits_softmax_kernel<<<dim3(NRES / 16, NH, 1), 256, 0, stream>>>(
      Qe, Ke, Qn2, Kn2, attn2d, pwbuf, mask, attn);

  // 5) attn @ [v_scalar | v_point] per head (batched over z = heads)
  gemm_wmma<<<dim3(NRES / 64, 3, NH), 128, 0, stream>>>(
      attn, NRES, (long long)NRES * NRES,
      Vc, 48, (long long)NRES * 48,
      nullptr,
      resE, 48, (long long)NRES * 48,
      NRES, 48, NRES);

  // 6) attn_over_2d (second 302 MB pass over inputs_2d) -> final_act[:,576:]
  attn_over_2d_kernel<<<dim3(NRES, 1, 1), 256, 0, stream>>>(attn, in2d, final_act);

  // 7) inverse frames, dist, concat -> final_act[:, :576]
  assemble_kernel<<<(NRES * NH + 255) / 256, 256, 0, stream>>>(
      resE, rot, trans, final_act);

  // 8) output projection (K=2112)
  gemm_wmma<<<dim3(NRES / 64, OUTD / 16, 1), 128, 0, stream>>>(
      final_act, FDIM, 0, wout, OUTD, 0, bout, out, OUTD, 0, NRES, OUTD, FDIM);
}
